// GCN_30880814859094
// MI455X (gfx1250) — matrix operand order repaired and checked by reading, not compiled, not run on hardware
//
#include <hip/hip_runtime.h>

typedef __attribute__((ext_vector_type(16))) _Float16 v16h;
typedef __attribute__((ext_vector_type(8)))  _Float16 v8h;
typedef __attribute__((ext_vector_type(8)))  float    v8f;

#define TPB 256

// ---------------------------------------------------------------------------
// Degree / normalization
// ---------------------------------------------------------------------------
__global__ void deg_init_kernel(float* __restrict__ deg, int n) {
    int i = blockIdx.x * blockDim.x + threadIdx.x;
    if (i < n) deg[i] = 1.0f;  // self-loop contributes 1 to every node's degree
}

__global__ void deg_edge_kernel(const long long* __restrict__ dst,
                                float* __restrict__ deg, int E) {
    int e = blockIdx.x * blockDim.x + threadIdx.x;
    if (e >= E) return;
    if (e + 8192 < E) __builtin_prefetch(&dst[e + 8192], 0, 0);
    atomicAdd(&deg[(int)dst[e]], 1.0f);
}

__global__ void dinv_kernel(const float* __restrict__ deg,
                            float* __restrict__ dinv, int n) {
    int i = blockIdx.x * blockDim.x + threadIdx.x;
    if (i >= n) return;
    float d = deg[i];
    dinv[i] = (d > 0.0f) ? rsqrtf(d) : 0.0f;
}

// ---------------------------------------------------------------------------
// Layer-1 transform: t1 = x @ W1  (M=100000, K=128, N=4 padded to 16)
// One wave per 16-row tile; 4x v_wmma_f32_16x16x32_f16 per tile.
// W1 is staged in LDS, transposed + f16, so B fragments come from two
// contiguous ds_load_b128's per lane — no divergent global loads.
//   A (16-bit 16x32): lane L -> row M=L%16; lanes 0-15: K={0..7,16..23},
//                     lanes 16-31: K={8..15,24..31}.
//   B (16-bit 32x16): lane L -> col N=L%16; lanes 0-15: K={0..7,16..23},
//                     lanes 16-31: K={8..15,24..31}.
//   D (f32 16x16):    VGPR r: lanes 0-15 -> M=r, lanes 16-31 -> M=8+r.
// ---------------------------------------------------------------------------
__global__ __launch_bounds__(TPB) void xw1_wmma_kernel(
    const float* __restrict__ x, const float* __restrict__ W1,
    float* __restrict__ t1, int ntiles) {
    __shared__ _Float16 wT[4][128];   // wT[col][k] = (f16) W1[k*4 + col]

    for (int idx = threadIdx.x; idx < 512; idx += TPB) {
        int k = idx & 127;
        int c = idx >> 7;
        wT[c][k] = (_Float16)W1[k * 4 + c];
    }
    __syncthreads();

    int wave = (int)((blockIdx.x * blockDim.x + threadIdx.x) >> 5);
    int lane = (int)(threadIdx.x & 31);
    if (wave >= ntiles) return;        // wave-uniform exit: EXEC all-1 at wmma

    const int row0 = wave * 16;
    const int half = lane >> 4;        // 0 or 1
    const int lrow = lane & 15;        // A: row, B: col, D: N
    const int col  = lrow & 3;         // clamped column (valid LDS address)
    const _Float16 bmask = (lrow < 4) ? (_Float16)1.0f : (_Float16)0.0f;

    v8f acc = {};
#pragma unroll
    for (int k0 = 0; k0 < 128; k0 += 32) {
        // ---- A: two 8-float strips of one x row, f32 -> f16 in-register
        const float4* xv =
            (const float4*)(x + (size_t)(row0 + lrow) * 128 + k0 + half * 8);
        float4 p0 = xv[0];             // K = k0 + half*8 + {0..3}
        float4 p1 = xv[1];             // K = k0 + half*8 + {4..7}
        float4 p2 = xv[4];             // K = k0 + 16 + half*8 + {0..3}
        float4 p3 = xv[5];             // K = k0 + 16 + half*8 + {4..7}
        v16h a;
        a[0]  = (_Float16)p0.x; a[1]  = (_Float16)p0.y;
        a[2]  = (_Float16)p0.z; a[3]  = (_Float16)p0.w;
        a[4]  = (_Float16)p1.x; a[5]  = (_Float16)p1.y;
        a[6]  = (_Float16)p1.z; a[7]  = (_Float16)p1.w;
        a[8]  = (_Float16)p2.x; a[9]  = (_Float16)p2.y;
        a[10] = (_Float16)p2.z; a[11] = (_Float16)p2.w;
        a[12] = (_Float16)p3.x; a[13] = (_Float16)p3.y;
        a[14] = (_Float16)p3.z; a[15] = (_Float16)p3.w;

        // ---- B: two 8-f16 runs from LDS, 16 k apart, masked for padded columns
        const v8h* wp = (const v8h*)&wT[col][k0 + half * 8];
        v8h w0 = wp[0];
        v8h w1 = wp[2];
        v16h b;
#pragma unroll
        for (int j = 0; j < 8; ++j) {
            b[j]     = bmask * w0[j];
            b[j + 8] = bmask * w1[j];
        }

        acc = __builtin_amdgcn_wmma_f32_16x16x32_f16(
            /*neg_a=*/false, a, /*neg_b=*/false, b,
            /*c_mod=*/(short)0, acc, /*reuse_a=*/false, /*reuse_b=*/false);
    }
    // ---- store D: only N (=lrow) < 4 columns are real
    if (lrow < 4) {
#pragma unroll
        for (int r = 0; r < 8; ++r) {
            int m = half * 8 + r;
            t1[(size_t)(row0 + m) * 4 + lrow] = acc[r];
        }
    }
}

// ---------------------------------------------------------------------------
// Aggregation: agg[i] = dinv[i]^2 * t[i]  (self loop), then edge scatter-adds
// ---------------------------------------------------------------------------
__global__ void agg_init4_kernel(const float* __restrict__ t,
                                 const float* __restrict__ dinv,
                                 float* __restrict__ agg, int n) {
    int i = blockIdx.x * blockDim.x + threadIdx.x;
    if (i >= n) return;
    float w = dinv[i] * dinv[i];
    float4 v = ((const float4*)t)[i];
    float4 r = make_float4(w * v.x, w * v.y, w * v.z, w * v.w);
    ((float4*)agg)[i] = r;
}

__global__ void agg_init2_kernel(const float* __restrict__ t,
                                 const float* __restrict__ dinv,
                                 float* __restrict__ agg, int n) {
    int i = blockIdx.x * blockDim.x + threadIdx.x;
    if (i >= n) return;
    float w = dinv[i] * dinv[i];
    float2 v = ((const float2*)t)[i];
    ((float2*)agg)[i] = make_float2(w * v.x, w * v.y);
}

__global__ void agg_edge4_kernel(const long long* __restrict__ src,
                                 const long long* __restrict__ dst,
                                 const float* __restrict__ dinv,
                                 const float* __restrict__ t,
                                 float* __restrict__ agg, int E) {
    int e = blockIdx.x * blockDim.x + threadIdx.x;
    if (e >= E) return;
    if (e + 8192 < E) {
        __builtin_prefetch(&src[e + 8192], 0, 0);
        __builtin_prefetch(&dst[e + 8192], 0, 0);
    }
    int s = (int)src[e];
    int d = (int)dst[e];
    float w = dinv[s] * dinv[d];
    float4 v = ((const float4*)t)[s];          // one b128 gather
    float* ap = agg + (size_t)d * 4;
    atomicAdd(ap + 0, w * v.x);
    atomicAdd(ap + 1, w * v.y);
    atomicAdd(ap + 2, w * v.z);
    atomicAdd(ap + 3, w * v.w);
}

__global__ void agg_edge2_kernel(const long long* __restrict__ src,
                                 const long long* __restrict__ dst,
                                 const float* __restrict__ dinv,
                                 const float* __restrict__ t,
                                 float* __restrict__ agg, int E) {
    int e = blockIdx.x * blockDim.x + threadIdx.x;
    if (e >= E) return;
    if (e + 8192 < E) {
        __builtin_prefetch(&src[e + 8192], 0, 0);
        __builtin_prefetch(&dst[e + 8192], 0, 0);
    }
    int s = (int)src[e];
    int d = (int)dst[e];
    float w = dinv[s] * dinv[d];
    float2 v = ((const float2*)t)[s];          // one b64 gather
    float* ap = agg + (size_t)d * 2;
    atomicAdd(ap + 0, w * v.x);
    atomicAdd(ap + 1, w * v.y);
}

// ---------------------------------------------------------------------------
// Fused PReLU epilogue + next-layer transform: t_next = prelu(agg+b, a) @ W
// (h1/h2 are never materialized)
// ---------------------------------------------------------------------------
template <int DI, int DO>
__global__ void prelu_dense_kernel(const float* __restrict__ agg,
                                   const float* __restrict__ b,
                                   const float* __restrict__ a,
                                   const float* __restrict__ W,
                                   float* __restrict__ t, int n) {
    int i = blockIdx.x * blockDim.x + threadIdx.x;
    if (i >= n) return;
    float h[DI];
#pragma unroll
    for (int k = 0; k < DI; ++k) {
        float v = agg[(size_t)i * DI + k] + b[k];
        h[k] = (v >= 0.0f) ? v : a[k] * v;
    }
#pragma unroll
    for (int o = 0; o < DO; ++o) {
        float acc = 0.0f;
#pragma unroll
        for (int k = 0; k < DI; ++k) acc += h[k] * W[k * DO + o];
        t[(size_t)i * DO + o] = acc;
    }
}

// Final fusion: h3 = prelu(agg+b3, a3);  out = h3 @ Wl + bl
__global__ void final_kernel(const float* __restrict__ agg,
                             const float* __restrict__ b3,
                             const float* __restrict__ a3,
                             const float* __restrict__ Wl,
                             const float* __restrict__ bl,
                             float* __restrict__ out,
                             float* __restrict__ h3, int n) {
    int i = blockIdx.x * blockDim.x + threadIdx.x;
    if (i >= n) return;
    float2 g = ((const float2*)agg)[i];
    float v0 = g.x + b3[0];
    float v1 = g.y + b3[1];
    float h0 = (v0 >= 0.0f) ? v0 : a3[0] * v0;
    float h1 = (v1 >= 0.0f) ? v1 : a3[1] * v1;
    ((float2*)h3)[i] = make_float2(h0, h1);
    float o0 = h0 * Wl[0] + h1 * Wl[2] + bl[0];
    float o1 = h0 * Wl[1] + h1 * Wl[3] + bl[1];
    ((float2*)out)[i] = make_float2(o0, o1);
}

// ---------------------------------------------------------------------------
extern "C" void kernel_launch(void* const* d_in, const int* in_sizes, int n_in,
                              void* d_out, int out_size, void* d_ws, size_t ws_size,
                              hipStream_t stream) {
    const float*     x  = (const float*)d_in[0];
    const long long* ei = (const long long*)d_in[1];   // int64 edge_index [2, E]
    const float* W1 = (const float*)d_in[2];
    const float* b1 = (const float*)d_in[3];
    const float* a1 = (const float*)d_in[4];
    const float* W2 = (const float*)d_in[5];
    const float* b2 = (const float*)d_in[6];
    const float* a2 = (const float*)d_in[7];
    const float* W3 = (const float*)d_in[8];
    const float* b3 = (const float*)d_in[9];
    const float* a3 = (const float*)d_in[10];
    const float* Wl = (const float*)d_in[11];
    const float* bl = (const float*)d_in[12];

    const int n = in_sizes[0] / 128;   // 100000 nodes
    const int E = in_sizes[1] / 2;     // 3200000 edges
    const long long* src = ei;
    const long long* dst = ei + (size_t)E;

    // workspace carve (floats): deg n | dinv n | tbuf 4n | agg 4n   (= 10n)
    float* ws   = (float*)d_ws;
    float* deg  = ws;
    float* dinv = ws + (size_t)n;
    float* tbuf = ws + (size_t)2 * n;
    float* agg  = ws + (size_t)6 * n;

    float* out = (float*)d_out;             // [n,2]
    float* h3  = out + (size_t)2 * n;       // [n,2]  (tuple's second output)

    auto nb = [](int m) { return (m + TPB - 1) / TPB; };

    // ---- degrees + D^{-1/2}
    deg_init_kernel<<<nb(n), TPB, 0, stream>>>(deg, n);
    deg_edge_kernel<<<nb(E), TPB, 0, stream>>>(dst, deg, E);
    dinv_kernel<<<nb(n), TPB, 0, stream>>>(deg, dinv, n);

    // ---- layer 1: WMMA transform, then aggregate
    const int ntiles = (n + 15) / 16;                 // 6250 (n % 16 == 0)
    const int wblocks = (ntiles * 32 + TPB - 1) / TPB;
    xw1_wmma_kernel<<<wblocks, TPB, 0, stream>>>(x, W1, tbuf, ntiles);
    agg_init4_kernel<<<nb(n), TPB, 0, stream>>>(tbuf, dinv, agg, n);
    agg_edge4_kernel<<<nb(E), TPB, 0, stream>>>(src, dst, dinv, tbuf, agg, E);

    // ---- layer 2: fused prelu+dense, then aggregate
    prelu_dense_kernel<4, 4><<<nb(n), TPB, 0, stream>>>(agg, b1, a1, W2, tbuf, n);
    agg_init4_kernel<<<nb(n), TPB, 0, stream>>>(tbuf, dinv, agg, n);
    agg_edge4_kernel<<<nb(E), TPB, 0, stream>>>(src, dst, dinv, tbuf, agg, E);

    // ---- layer 3: fused prelu+dense, then aggregate
    prelu_dense_kernel<4, 2><<<nb(n), TPB, 0, stream>>>(agg, b2, a2, W3, tbuf, n);
    agg_init2_kernel<<<nb(n), TPB, 0, stream>>>(tbuf, dinv, agg, n);
    agg_edge2_kernel<<<nb(E), TPB, 0, stream>>>(src, dst, dinv, tbuf, agg, E);

    // ---- fused PReLU + final linear, writes both tuple outputs
    final_kernel<<<nb(n), TPB, 0, stream>>>(agg, b3, a3, Wl, bl, out, h3, n);
}
